// JiTFluidDiffusion_57458072486177
// MI455X (gfx1250) — matrix operand order repaired and checked by reading, not compiled, not run on hardware
//
#include <hip/hip_runtime.h>
#include <hip/hip_bf16.h>

typedef __attribute__((ext_vector_type(16))) __bf16 v16bf;
typedef __attribute__((ext_vector_type(8)))  __bf16 v8bf;
typedef __attribute__((ext_vector_type(8)))  float  v8f;

typedef int v4i_vs __attribute__((vector_size(16)));
typedef __attribute__((address_space(1))) v4i_vs* as1_v4i;
typedef __attribute__((address_space(3))) v4i_vs* as3_v4i;

#define D_MODEL 768
#define NTOK    1024
#define BATCH   2
#define NHEAD   12
#define HDIM    64
#define D3      2304
#define FFH     2048
#define ADA     4608
#define NLAYER  12

#if defined(__has_builtin)
#if __has_builtin(__builtin_amdgcn_global_load_async_to_lds_b128) && \
    __has_builtin(__builtin_amdgcn_s_wait_asynccnt)
#define USE_ASYNC 1
#endif
#endif
#ifndef USE_ASYNC
#define USE_ASYNC 0
#endif

__device__ __forceinline__ v8f wmma_bf16(v16bf a, v16bf b, v8f c) {
    return __builtin_amdgcn_wmma_f32_16x16x32_bf16(false, a, false, b, (short)0, c, false, false);
}

__device__ __forceinline__ v16bf make_afrag(const __bf16* lo_p, const __bf16* hi_p) {
    v8bf lo = *(const v8bf*)lo_p;
    v8bf hi = *(const v8bf*)hi_p;
    return __builtin_shufflevector(lo, hi, 0,1,2,3,4,5,6,7,8,9,10,11,12,13,14,15);
}

// copy 8 bf16 (16 bytes) global -> LDS, async when available
__device__ __forceinline__ void copy16(__bf16* ldsdst, const __bf16* gsrc) {
#if USE_ASYNC
    __builtin_amdgcn_global_load_async_to_lds_b128(
        (as1_v4i)gsrc, (as3_v4i)ldsdst, 0, 0);
#else
    *(v8bf*)ldsdst = *(const v8bf*)gsrc;
#endif
}
__device__ __forceinline__ void wait_async() {
#if USE_ASYNC
    __builtin_amdgcn_s_wait_asynccnt(0);
#endif
}

// ---------------------------------------------------------------------------
// WMMA GEMM: out[M,Nout] = A[M,K](bf16) @ W[Nout,K](bf16)^T (+bias)
// mode 0: f32 out = acc + bias
// mode 1: f32 out = resid + gate[b, gateOff + c] * (acc + bias)
// mode 2: bf16 out = acc + bias
// K multiple of 32; M multiple of 128. Double-buffered LDS, async staging.
// ---------------------------------------------------------------------------
__global__ __launch_bounds__(256)
void gemm_wmma(const __bf16* __restrict__ A, const __bf16* __restrict__ W,
               const float* __restrict__ bias, void* __restrict__ outp,
               int M, int Nout, int K, int mode,
               const float* __restrict__ resid, const float* __restrict__ gate,
               int gateStride, int gateOff)
{
    __shared__ __align__(64) __bf16 As[2][128][32];
    __shared__ __align__(64) __bf16 Ws[2][128][32];

    const int tid  = threadIdx.x;
    const int wid  = tid >> 5;
    const int lane = tid & 31;
    const int rowBase = blockIdx.y * 128;
    const int colBase = blockIdx.x * 128;
    const int wr = (wid >> 2) * 64;
    const int wc = (wid & 3) * 32;

    // loader: clamp rows instead of predicating (garbage is never stored)
    const int lrow = tid >> 1;
    const int lk   = (tid & 1) * 16;
    const int gr = min(rowBase + lrow, M - 1);
    const int gn = min(colBase + lrow, Nout - 1);
    const __bf16* Arow = A + (size_t)gr * K;
    const __bf16* Wrow = W + (size_t)gn * K;

    const v8f vz = {0.f,0.f,0.f,0.f,0.f,0.f,0.f,0.f};
    v8f acc[4][2];
    for (int i = 0; i < 4; ++i) for (int j = 0; j < 2; ++j) acc[i][j] = vz;

    auto issue = [&](int k0, int bufi) {
        copy16(&As[bufi][lrow][lk],     Arow + k0 + lk);
        copy16(&As[bufi][lrow][lk + 8], Arow + k0 + lk + 8);
        copy16(&Ws[bufi][lrow][lk],     Wrow + k0 + lk);
        copy16(&Ws[bufi][lrow][lk + 8], Wrow + k0 + lk + 8);
    };

    issue(0, 0);
    wait_async();
    __syncthreads();

    const int m   = lane & 15;
    const int kbA = (lane >> 4) * 16;
    const int kbB = (lane >> 4) * 16;

    int bufi = 0;
    for (int k0 = 0; k0 < K; k0 += 32, bufi ^= 1) {
        const bool hasNext = (k0 + 32) < K;
        if (hasNext) issue(k0 + 32, bufi ^ 1);

        v16bf a[4];
        #pragma unroll
        for (int i = 0; i < 4; ++i)
            a[i] = make_afrag(&As[bufi][wr + 16*i + m][kbA],
                              &As[bufi][wr + 16*i + m][kbA + 8]);
        v16bf b[2];
        #pragma unroll
        for (int j = 0; j < 2; ++j)
            b[j] = *(const v16bf*)&Ws[bufi][wc + 16*j + m][kbB];
        #pragma unroll
        for (int i = 0; i < 4; ++i)
            #pragma unroll
            for (int j = 0; j < 2; ++j)
                acc[i][j] = wmma_bf16(a[i], b[j], acc[i][j]);

        if (hasNext) wait_async();
        __syncthreads();
    }

    const int halo = (lane >> 4) * 8;
    #pragma unroll
    for (int i = 0; i < 4; ++i)
        for (int j = 0; j < 2; ++j)
            #pragma unroll
            for (int jv = 0; jv < 8; ++jv) {
                int r = rowBase + wr + 16*i + jv + halo;
                int c = colBase + wc + 16*j + m;
                if (r < M && c < Nout) {
                    float v = acc[i][j][jv] + (bias ? bias[c] : 0.f);
                    if (mode == 1) {
                        int bb = r / NTOK;
                        v = resid[(size_t)r * Nout + c] +
                            gate[bb * gateStride + gateOff + c] * v;
                    }
                    if (mode == 2) ((__bf16*)outp)[(size_t)r * Nout + c] = (__bf16)v;
                    else           ((float*)outp)[(size_t)r * Nout + c] = v;
                }
            }
}

// ---------------------------------------------------------------------------
// Flash attention, one (b,h,qtile=128) per block. qkv fp32 [2048,2304];
// writes bf16 o [2048,768].
// ---------------------------------------------------------------------------
__global__ __launch_bounds__(256)
void flash_attn(const float* __restrict__ qkv, __bf16* __restrict__ o)
{
    const int bh = blockIdx.x;
    const int b  = bh / NHEAD;
    const int h  = bh % NHEAD;
    const int qbase = blockIdx.y * 128;
    const int tid = threadIdx.x, wid = tid >> 5, lane = tid & 31;

    __shared__ __align__(64) __bf16 Qs[128][64];
    __shared__ __align__(64) __bf16 Ks[128][64];
    __shared__ __align__(64) __bf16 Vt[64][128];
    __shared__ __align__(64) __bf16 Ps[8][16][128];

    const float* qptr = qkv + (size_t)(b * NTOK) * D3 + h * HDIM;
    const float* kptr = qptr + D_MODEL;
    const float* vptr = qptr + 2 * D_MODEL;

    {   // load Q tile (128 x 64)
        const int r  = tid >> 1;
        const int c0 = (tid & 1) * 32;
        const float4* s4 = (const float4*)(qptr + (size_t)(qbase + r) * D3 + c0);
        #pragma unroll
        for (int q = 0; q < 8; ++q) {
            float4 v = s4[q];
            Qs[r][c0 + 4*q + 0] = (__bf16)v.x;
            Qs[r][c0 + 4*q + 1] = (__bf16)v.y;
            Qs[r][c0 + 4*q + 2] = (__bf16)v.z;
            Qs[r][c0 + 4*q + 3] = (__bf16)v.w;
        }
    }

    float mstate[8], lstate[8];
    #pragma unroll
    for (int jv = 0; jv < 8; ++jv) { mstate[jv] = -1e30f; lstate[jv] = 0.f; }
    const v8f vz = {0.f,0.f,0.f,0.f,0.f,0.f,0.f,0.f};
    v8f oacc[4];
    for (int j = 0; j < 4; ++j) oacc[j] = vz;

    const int m    = lane & 15;
    const int kbA  = (lane >> 4) * 16;
    const int kbB  = (lane >> 4) * 16;
    const int halo = (lane >> 4) * 8;

    for (int kt = 0; kt < NTOK; kt += 128) {
        __syncthreads();
        {
            const int r  = tid >> 1;
            const int c0 = (tid & 1) * 32;
            const float4* sk = (const float4*)(kptr + (size_t)(kt + r) * D3 + c0);
            const float4* sv = (const float4*)(vptr + (size_t)(kt + r) * D3 + c0);
            #pragma unroll
            for (int q = 0; q < 8; ++q) {
                float4 kv = sk[q];
                Ks[r][c0 + 4*q + 0] = (__bf16)kv.x;
                Ks[r][c0 + 4*q + 1] = (__bf16)kv.y;
                Ks[r][c0 + 4*q + 2] = (__bf16)kv.z;
                Ks[r][c0 + 4*q + 3] = (__bf16)kv.w;
                float4 vv = sv[q];
                Vt[c0 + 4*q + 0][r] = (__bf16)vv.x;
                Vt[c0 + 4*q + 1][r] = (__bf16)vv.y;
                Vt[c0 + 4*q + 2][r] = (__bf16)vv.z;
                Vt[c0 + 4*q + 3][r] = (__bf16)vv.w;
            }
        }
        __syncthreads();

        v8f s[8];
        for (int j = 0; j < 8; ++j) s[j] = vz;
        #pragma unroll
        for (int ks = 0; ks < 64; ks += 32) {
            v16bf a = make_afrag(&Qs[16*wid + m][ks + kbA], &Qs[16*wid + m][ks + kbA + 8]);
            #pragma unroll
            for (int j = 0; j < 8; ++j) {
                v16bf bf = *(const v16bf*)&Ks[16*j + m][ks + kbB];
                s[j] = wmma_bf16(a, bf, s[j]);
            }
        }

        #pragma unroll
        for (int jv = 0; jv < 8; ++jv) {
            float mx = -1e30f;
            #pragma unroll
            for (int j = 0; j < 8; ++j) { s[j][jv] *= 0.125f; mx = fmaxf(mx, s[j][jv]); }
            #pragma unroll
            for (int off = 1; off < 16; off <<= 1) mx = fmaxf(mx, __shfl_xor(mx, off, 32));
            float mnew  = fmaxf(mstate[jv], mx);
            float alpha = __expf(mstate[jv] - mnew);
            mstate[jv] = mnew;
            float rsum = 0.f;
            #pragma unroll
            for (int j = 0; j < 8; ++j) {
                float p = __expf(s[j][jv] - mnew);
                s[j][jv] = p;
                rsum += p;
            }
            #pragma unroll
            for (int off = 1; off < 16; off <<= 1) rsum += __shfl_xor(rsum, off, 32);
            lstate[jv] = lstate[jv] * alpha + rsum;
            #pragma unroll
            for (int j = 0; j < 4; ++j) oacc[j][jv] *= alpha;
            #pragma unroll
            for (int j = 0; j < 8; ++j)
                Ps[wid][jv + halo][16*j + m] = (__bf16)s[j][jv];
        }

        #pragma unroll
        for (int ks = 0; ks < 128; ks += 32) {
            v16bf a = make_afrag(&Ps[wid][m][ks + kbA], &Ps[wid][m][ks + kbA + 8]);
            #pragma unroll
            for (int j = 0; j < 4; ++j) {
                v16bf bf = *(const v16bf*)&Vt[16*j + m][ks + kbB];
                oacc[j] = wmma_bf16(a, bf, oacc[j]);
            }
        }
    }

    #pragma unroll
    for (int jv = 0; jv < 8; ++jv) {
        float inv = 1.f / lstate[jv];
        int r = qbase + 16*wid + jv + halo;
        size_t base = (size_t)(b * NTOK + r) * D_MODEL + h * HDIM;
        #pragma unroll
        for (int j = 0; j < 4; ++j)
            o[base + 16*j + m] = (__bf16)(oacc[j][jv] * inv);
    }
}

// ---------------------------------------------------------------------------
__global__ __launch_bounds__(256)
void gemv_k(const float* __restrict__ x, const float* __restrict__ W,
            const float* __restrict__ bias, float* __restrict__ out,
            int Bn, int Nout, int K, int act)
{
    int gw = (int)((blockIdx.x * blockDim.x + threadIdx.x) >> 5);
    int lane = threadIdx.x & 31;
    if (gw >= Bn * Nout) return;
    int n = gw % Nout, b = gw / Nout;
    const float* xr = x + (size_t)b * K;
    const float* wr = W + (size_t)n * K;
    float s = 0.f;
    for (int k = lane; k < K; k += 32) s += xr[k] * wr[k];
    #pragma unroll
    for (int off = 16; off; off >>= 1) s += __shfl_xor(s, off, 32);
    if (lane == 0) {
        float v = s + (bias ? bias[n] : 0.f);
        if (act) v = v / (1.f + __expf(-v));
        out[(size_t)b * Nout + n] = v;
    }
}

// RMSNorm + adaLN mod, writes bf16 (GEMM A operand)
__global__ __launch_bounds__(256)
void rmsmod(const float* __restrict__ hbuf, const float* __restrict__ w,
            const float* __restrict__ mbuf, int mStride, int shOff, int scOff,
            __bf16* __restrict__ z)
{
    const int r = blockIdx.x;
    const int b = r / NTOK;
    const float* row = hbuf + (size_t)r * D_MODEL;
    __shared__ float red[8];
    float ss = 0.f;
    for (int d = threadIdx.x; d < D_MODEL; d += 256) { float v = row[d]; ss += v * v; }
    #pragma unroll
    for (int off = 16; off; off >>= 1) ss += __shfl_down(ss, off, 32);
    if ((threadIdx.x & 31) == 0) red[threadIdx.x >> 5] = ss;
    __syncthreads();
    float sum = 0.f;
    #pragma unroll
    for (int i = 0; i < 8; ++i) sum += red[i];
    float inv = rsqrtf(sum / (float)D_MODEL + 1e-6f);
    const float* mrow = mbuf + (size_t)b * mStride;
    for (int d = threadIdx.x; d < D_MODEL; d += 256) {
        float v = row[d] * inv * w[d];
        z[(size_t)r * D_MODEL + d] = (__bf16)(v * (1.f + mrow[scOff + d]) + mrow[shOff + d]);
    }
}

__global__ __launch_bounds__(256)
void qk_rms(float* __restrict__ qkv, const float* __restrict__ qn,
            const float* __restrict__ kn)
{
    int gw = (int)((blockIdx.x * blockDim.x + threadIdx.x) >> 5);
    int lane = threadIdx.x & 31;
    int s  = gw & 1;
    int hh = (gw >> 1) % NHEAD;
    int r  = gw / (2 * NHEAD);
    if (r >= BATCH * NTOK) return;
    float* p = qkv + (size_t)r * D3 + s * D_MODEL + hh * HDIM;
    float v0 = p[lane], v1 = p[lane + 32];
    float ss = v0 * v0 + v1 * v1;
    #pragma unroll
    for (int off = 16; off; off >>= 1) ss += __shfl_xor(ss, off, 32);
    float inv = rsqrtf(ss / 64.f + 1e-6f);
    const float* wv = s ? kn : qn;
    p[lane]      = v0 * inv * wv[lane];
    p[lane + 32] = v1 * inv * wv[lane + 32];
}

// ---------------------------------------------------------------------------
__global__ void cvt_bf16_k(const float* __restrict__ in, __bf16* __restrict__ out,
                           size_t n)
{
    size_t base = (size_t)blockIdx.x * 1024;
    #pragma unroll
    for (int q = 0; q < 4; ++q) {
        size_t j = base + q * 256 + threadIdx.x;
        if (j < n) out[j] = (__bf16)in[j];
    }
}

__global__ void patchify(const float* __restrict__ x, __bf16* __restrict__ p, int C)
{
    int i = blockIdx.x * blockDim.x + threadIdx.x;
    int K = C * 256;
    int total = BATCH * NTOK * K;
    if (i >= total) return;
    int f = i % K; int token = (i / K) % NTOK; int b = i / (K * NTOK);
    int c = f >> 8, py = (f >> 4) & 15, px = f & 15;
    int gy = token >> 5, gx = token & 31;
    p[i] = (__bf16)x[((size_t)(b * C + c) * 512 + gy * 16 + py) * 512 + gx * 16 + px];
}

__global__ void unpatchify(const float* __restrict__ outp, float* __restrict__ out)
{
    int i = blockIdx.x * blockDim.x + threadIdx.x;
    if (i >= BATCH * 512 * 512) return;
    int px_ = i % 512, py_ = (i / 512) % 512, b = i / (512 * 512);
    int gy = py_ >> 4, py = py_ & 15, gx = px_ >> 4, px = px_ & 15;
    out[i] = outp[(size_t)(b * NTOK + gy * 32 + gx) * 256 + py * 16 + px];
}

__global__ void add_posc(float* __restrict__ h, const float* __restrict__ pos,
                         const float* __restrict__ pec)
{
    int i = blockIdx.x * blockDim.x + threadIdx.x;
    if (i >= BATCH * NTOK * D_MODEL) return;
    h[i] = h[i] + pos[i % (NTOK * D_MODEL)] + pec[i];
}

__global__ void temb_init(const float* __restrict__ t, float* __restrict__ out)
{
    int i = blockIdx.x * blockDim.x + threadIdx.x;
    if (i >= BATCH * 256) return;
    int b = i >> 8, j = i & 255;
    float freq = __expf(-logf(10000.f) * (float)(j & 127) / 128.f);
    float arg  = t[b] * freq;
    out[i] = (j < 128) ? __cosf(arg) : __sinf(arg);
}

__global__ void concat_u(const float* __restrict__ u_past,
                         const float* __restrict__ u_curr, float* __restrict__ u)
{
    int i = blockIdx.x * blockDim.x + threadIdx.x;
    if (i >= BATCH * 11) return;
    int b = i / 11, j = i % 11;
    u[i] = (j < 10) ? u_past[b * 10 + j] : u_curr[b];
}

__global__ void add_silu(const float* __restrict__ a, const float* __restrict__ b,
                         float* __restrict__ o, int n)
{
    int i = blockIdx.x * blockDim.x + threadIdx.x;
    if (i >= n) return;
    float v = a[i] + b[i];
    o[i] = v / (1.f + __expf(-v));
}

__global__ void silu_gate(const float* __restrict__ x12, __bf16* __restrict__ xg)
{
    int i = blockIdx.x * blockDim.x + threadIdx.x;
    int total = BATCH * NTOK * FFH;
    if (i >= total) return;
    int r = i / FFH, j = i % FFH;
    float a  = x12[(size_t)r * 2 * FFH + j];
    float bb = x12[(size_t)r * 2 * FFH + FFH + j];
    float s  = a / (1.f + __expf(-a));
    xg[i] = (__bf16)(s * bb);
}

// ---------------------------------------------------------------------------
extern "C" void kernel_launch(void* const* d_in, const int* in_sizes, int n_in,
                              void* d_out, int out_size, void* d_ws, size_t ws_size,
                              hipStream_t stream)
{
    (void)in_sizes; (void)n_in; (void)out_size; (void)ws_size;
    const float* x      = (const float*)d_in[0];
    const float* x_past = (const float*)d_in[1];
    const float* u_past = (const float*)d_in[2];
    const float* u_curr = (const float*)d_in[3];
    const float* t_in   = (const float*)d_in[4];
    const float* pos    = (const float*)d_in[5];
    const float* xe_w1  = (const float*)d_in[6];
    const float* xe_w2  = (const float*)d_in[7];
    const float* xe_b2  = (const float*)d_in[8];
    const float* ce_w1  = (const float*)d_in[9];
    const float* ce_w2  = (const float*)d_in[10];
    const float* ce_b2  = (const float*)d_in[11];
    const float* t_w1   = (const float*)d_in[12];
    const float* t_b1   = (const float*)d_in[13];
    const float* t_w2   = (const float*)d_in[14];
    const float* t_b2   = (const float*)d_in[15];
    const float* c_w1   = (const float*)d_in[16];
    const float* c_b1   = (const float*)d_in[17];
    const float* c_w2   = (const float*)d_in[18];
    const float* c_b2   = (const float*)d_in[19];
    const float* n1_w   = (const float*)d_in[20];
    const float* qkv_w  = (const float*)d_in[21];
    const float* qkv_b  = (const float*)d_in[22];
    const float* qn_w   = (const float*)d_in[23];
    const float* kn_w   = (const float*)d_in[24];
    const float* ao_w   = (const float*)d_in[25];
    const float* ao_b   = (const float*)d_in[26];
    const float* n2_w   = (const float*)d_in[27];
    const float* w12_w  = (const float*)d_in[28];
    const float* w12_b  = (const float*)d_in[29];
    const float* w3_w   = (const float*)d_in[30];
    const float* w3_b   = (const float*)d_in[31];
    const float* ada_w  = (const float*)d_in[32];
    const float* ada_b  = (const float*)d_in[33];
    const float* fn_w   = (const float*)d_in[34];
    const float* fl_w   = (const float*)d_in[35];
    const float* fl_b   = (const float*)d_in[36];
    const float* fa_w   = (const float*)d_in[37];
    const float* fa_b   = (const float*)d_in[38];

    const int M = BATCH * NTOK; // 2048

    char* wsb = (char*)d_ws;
    size_t off = 0;
    auto allocB = [&](size_t bytes) {
        void* p = wsb + off; off += (bytes + 255) & ~(size_t)255; return p;
    };
    auto allocF  = [&](size_t n) { return (float*)allocB(n * 4); };
    auto allocBF = [&](size_t n) { return (__bf16*)allocB(n * 2); };

    // bf16 activations
    __bf16* p_x  = allocBF((size_t)M * 256);
    __bf16* p_c  = allocBF((size_t)M * 2560);
    __bf16* e1   = allocBF((size_t)M * 64);
    __bf16* zbuf = allocBF((size_t)M * D_MODEL);
    __bf16* obuf = allocBF((size_t)M * D_MODEL);
    __bf16* xg   = allocBF((size_t)M * FFH);
    // fp32 activations
    float* hbuf = allocF((size_t)M * D_MODEL);
    float* pec  = allocF((size_t)M * D_MODEL);
    float* qkvb = allocF((size_t)M * D3);
    float* x12  = allocF((size_t)M * 2 * FFH);
    float* outp = allocF((size_t)M * 256);
    float* ti   = allocF(BATCH * 256);
    float* th   = allocF(BATCH * D_MODEL);
    float* temb = allocF(BATCH * D_MODEL);
    float* uu   = allocF(BATCH * 16);
    float* ch   = allocF(BATCH * D_MODEL);
    float* cemb = allocF(BATCH * D_MODEL);
    float* csb  = allocF(BATCH * D_MODEL);
    float* mbuf = allocF(BATCH * ADA);
    float* m2   = allocF(BATCH * 2 * D_MODEL);
    // bf16 weights (converted once per launch)
    __bf16* wqkv = allocBF((size_t)NLAYER * D3 * D_MODEL);
    __bf16* wao  = allocBF((size_t)NLAYER * D_MODEL * D_MODEL);
    __bf16* w12b = allocBF((size_t)NLAYER * 2 * FFH * D_MODEL);
    __bf16* w3b  = allocBF((size_t)NLAYER * D_MODEL * FFH);
    __bf16* wxe1 = allocBF((size_t)64 * 256);
    __bf16* wxe2 = allocBF((size_t)D_MODEL * 64);
    __bf16* wce1 = allocBF((size_t)64 * 2560);
    __bf16* wce2 = allocBF((size_t)D_MODEL * 64);
    __bf16* wfl  = allocBF((size_t)256 * D_MODEL);

    auto cvt = [&](const float* src, __bf16* dst, size_t n) {
        cvt_bf16_k<<<(unsigned)((n + 1023) / 1024), 256, 0, stream>>>(src, dst, n);
    };
    auto gemm = [&](const __bf16* A, const __bf16* W, const float* bias, void* out,
                    int Mm, int Nout, int K, int mode, const float* resid,
                    const float* gate, int gStride, int gOff) {
        dim3 grid((Nout + 127) / 128, (Mm + 127) / 128);
        gemm_wmma<<<grid, dim3(256), 0, stream>>>(A, W, bias, out, Mm, Nout, K,
                                                  mode, resid, gate, gStride, gOff);
    };
    auto gemv = [&](const float* xx, const float* W, const float* bias, float* out,
                    int Bn, int Nout, int K, int act) {
        int blocks = (Bn * Nout * 32 + 255) / 256;
        gemv_k<<<blocks, 256, 0, stream>>>(xx, W, bias, out, Bn, Nout, K, act);
    };
    auto ew = [&](int n) { return dim3((n + 255) / 256); };

    // ---- weight conversion (once per launch) ----
    cvt(qkv_w, wqkv, (size_t)NLAYER * D3 * D_MODEL);
    cvt(ao_w,  wao,  (size_t)NLAYER * D_MODEL * D_MODEL);
    cvt(w12_w, w12b, (size_t)NLAYER * 2 * FFH * D_MODEL);
    cvt(w3_w,  w3b,  (size_t)NLAYER * D_MODEL * FFH);
    cvt(xe_w1, wxe1, (size_t)64 * 256);
    cvt(xe_w2, wxe2, (size_t)D_MODEL * 64);
    cvt(ce_w1, wce1, (size_t)64 * 2560);
    cvt(ce_w2, wce2, (size_t)D_MODEL * 64);
    cvt(fl_w,  wfl,  (size_t)256 * D_MODEL);

    // ---- patch embeds ----
    patchify<<<ew(M * 256), 256, 0, stream>>>(x, p_x, 1);
    patchify<<<ew(M * 2560), 256, 0, stream>>>(x_past, p_c, 10);
    gemm(p_x, wxe1, nullptr, e1, M, 64, 256, 2, nullptr, nullptr, 0, 0);
    gemm(e1, wxe2, xe_b2, hbuf, M, D_MODEL, 64, 0, nullptr, nullptr, 0, 0);
    gemm(p_c, wce1, nullptr, e1, M, 64, 2560, 2, nullptr, nullptr, 0, 0);
    gemm(e1, wce2, ce_b2, pec, M, D_MODEL, 64, 0, nullptr, nullptr, 0, 0);
    add_posc<<<ew(M * D_MODEL), 256, 0, stream>>>(hbuf, pos, pec);

    // ---- conditioning ----
    temb_init<<<ew(BATCH * 256), 256, 0, stream>>>(t_in, ti);
    gemv(ti, t_w1, t_b1, th, BATCH, D_MODEL, 256, 1);
    gemv(th, t_w2, t_b2, temb, BATCH, D_MODEL, D_MODEL, 0);
    concat_u<<<1, 64, 0, stream>>>(u_past, u_curr, uu);
    gemv(uu, c_w1, c_b1, ch, BATCH, D_MODEL, 11, 1);
    gemv(ch, c_w2, c_b2, cemb, BATCH, D_MODEL, D_MODEL, 0);
    add_silu<<<ew(BATCH * D_MODEL), 256, 0, stream>>>(temb, cemb, csb, BATCH * D_MODEL);

    // ---- transformer layers ----
    for (int i = 0; i < NLAYER; ++i) {
        const __bf16* qkv_wi = wqkv + (size_t)i * D3 * D_MODEL;
        const __bf16* ao_wi  = wao  + (size_t)i * D_MODEL * D_MODEL;
        const __bf16* w12_wi = w12b + (size_t)i * 2 * FFH * D_MODEL;
        const __bf16* w3_wi  = w3b  + (size_t)i * D_MODEL * FFH;
        const float*  ada_wi = ada_w + (size_t)i * ADA * D_MODEL;

        gemv(csb, ada_wi, ada_b + (size_t)i * ADA, mbuf, BATCH, ADA, D_MODEL, 0);

        rmsmod<<<M, 256, 0, stream>>>(hbuf, n1_w + (size_t)i * D_MODEL, mbuf, ADA,
                                      0, D_MODEL, zbuf);
        gemm(zbuf, qkv_wi, qkv_b + (size_t)i * D3, qkvb, M, D3, D_MODEL,
             0, nullptr, nullptr, 0, 0);
        qk_rms<<<(M * 2 * NHEAD * 32 + 255) / 256, 256, 0, stream>>>(
            qkvb, qn_w + (size_t)i * HDIM, kn_w + (size_t)i * HDIM);
        flash_attn<<<dim3(BATCH * NHEAD, NTOK / 128), 256, 0, stream>>>(qkvb, obuf);
        gemm(obuf, ao_wi, ao_b + (size_t)i * D_MODEL, hbuf, M, D_MODEL, D_MODEL,
             1, hbuf, mbuf, ADA, 2 * D_MODEL);

        rmsmod<<<M, 256, 0, stream>>>(hbuf, n2_w + (size_t)i * D_MODEL, mbuf, ADA,
                                      3 * D_MODEL, 4 * D_MODEL, zbuf);
        gemm(zbuf, w12_wi, w12_b + (size_t)i * 2 * FFH, x12, M, 2 * FFH, D_MODEL,
             0, nullptr, nullptr, 0, 0);
        silu_gate<<<ew(M * FFH), 256, 0, stream>>>(x12, xg);
        gemm(xg, w3_wi, w3_b + (size_t)i * D_MODEL, hbuf, M, D_MODEL, FFH,
             1, hbuf, mbuf, ADA, 5 * D_MODEL);
    }

    // ---- final layer ----
    gemv(csb, fa_w, fa_b, m2, BATCH, 2 * D_MODEL, D_MODEL, 0);
    rmsmod<<<M, 256, 0, stream>>>(hbuf, fn_w, m2, 2 * D_MODEL, 0, D_MODEL, zbuf);
    gemm(zbuf, wfl, fl_b, outp, M, 256, D_MODEL, 0, nullptr, nullptr, 0, 0);
    unpatchify<<<ew(BATCH * 512 * 512), 256, 0, stream>>>(outp, (float*)d_out);
}
